// NoisyTopKRouter_21114059227288
// MI455X (gfx1250) — matrix operand order repaired and checked by reading, not compiled, not run on hardware
//
#include <hip/hip_runtime.h>
#include <hip/hip_bf16.h>
#include <math.h>

// ---------------------------------------------------------------------------
// MoE router (Switch-style, eval path) for MI455X / gfx1250, wave32 + WMMA.
//
// Roofline: GEMM [16384,4096]x[4096,64] = 8.6 GFLOP, streams x once (268 MB)
// -> 11.5 us @ 23.3 TB/s. fp32 is emulated as split-bf16 (hi+lo, 3 products
// of v_wmma_f32_16x16x32_bf16) -> 8/3x deeper K per instruction than fp32
// WMMA, keeping the kernel HBM-bound. W is pre-packed ONCE per launch into
// the exact wave32 B-fragment layout (bf16 hi/lo, 1 MB total, L2-resident),
// so the hot loop has no W conversions, no LDS, no barriers: just x-stream
// loads, a cheap truncating A split, and 12 WMMAs per 32-wide K chunk.
// ---------------------------------------------------------------------------

#define D_MODEL 4096
#define N_EXP   64
#define TOPK    2
#define KCHUNKS (D_MODEL / 32)                  // 128
#define WFRAG_N (KCHUNKS * 4 * 32 * 16)         // bf16 elems per W array
#define LC      7680                            // pass-2 LDS compaction cap

typedef __attribute__((ext_vector_type(8)))  float    v8f;
typedef __attribute__((ext_vector_type(8)))  unsigned v8u;
typedef __attribute__((ext_vector_type(16))) __bf16   v16bf;

// W pre-packed into wave32 B-fragment layout: [kc][tile][lane][e]
__device__ unsigned short g_Whi[WFRAG_N];       // 512 KB
__device__ unsigned short g_Wlo[WFRAG_N];       // 512 KB

static __device__ __forceinline__ unsigned short f32_to_bf16_rn(float f) {
    unsigned u = __float_as_uint(f);
    unsigned r = u + 0x7FFFu + ((u >> 16) & 1u);   // round-to-nearest-even
    return (unsigned short)(r >> 16);
}
static __device__ __forceinline__ float bf16_to_f32(unsigned short h) {
    return __uint_as_float(((unsigned)h) << 16);
}
static __device__ __forceinline__ bool beat(float a, int ea, float b, int eb) {
    return (a > b) || (a == b && ea < eb);   // jnp.top_k tie-break: lower index
}

// ---------------------------------------------------------------------------
// K_pre: pack W fp32 -> bf16 hi/lo directly in B-fragment order.
//   fragment value = W[t*16 + (lane&15)][kc*32 + (e>>3)*16 + (lane>>4)*8 + (e&7)]
// ---------------------------------------------------------------------------
__global__ __launch_bounds__(256) void moe_wpack_kernel(const float* __restrict__ W) {
    int i = blockIdx.x * 256 + threadIdx.x;
    if (i >= WFRAG_N) return;
    int e    = i & 15;
    int lane = (i >> 4) & 31;
    int t    = (i >> 9) & 3;
    int kc   = i >> 11;
    int row  = t * 16 + (lane & 15);
    int col  = kc * 32 + (e >> 3) * 16 + (lane >> 4) * 8 + (e & 7);
    float f = W[(size_t)row * D_MODEL + col];
    unsigned short h = f32_to_bf16_rn(f);
    g_Whi[i] = h;
    g_Wlo[i] = f32_to_bf16_rn(f - bf16_to_f32(h));
}

// ---------------------------------------------------------------------------
// K0: zero the mask region of d_out and the ws accumulators.
// ---------------------------------------------------------------------------
__global__ void moe_init_kernel(float* __restrict__ mask, float* __restrict__ ws, int NK) {
    int i = blockIdx.x * 256 + threadIdx.x;
    if (i < NK) mask[i] = 0.0f;
    if (blockIdx.x == 0 && threadIdx.x < 2 * N_EXP) ws[threadIdx.x] = 0.0f;
}

// ---------------------------------------------------------------------------
// K1: router GEMM (split-bf16 WMMA) + softmax + top-2 + importance atomics.
// Block = 256 threads = 8 waves; wave handles 16 tokens x 64 experts.
// No LDS, no barriers: B fragments come pre-packed from L2.
// ---------------------------------------------------------------------------
__global__ __launch_bounds__(256) void moe_router_gemm_kernel(
    const float* __restrict__ x, const float* __restrict__ b,
    float* __restrict__ out_ids, float* __restrict__ out_gates,
    float* __restrict__ imp, int N)
{
    const int tid  = threadIdx.x;
    const int wave = tid >> 5;
    const int lane = tid & 31;
    const int half = lane >> 4;     // wave32: two 16-lane halves
    const int ln   = lane & 15;
    const int tb   = blockIdx.x * 128 + wave * 16;   // this wave's 16 tokens

    const v16bf* __restrict__ wh = reinterpret_cast<const v16bf*>(g_Whi);
    const v16bf* __restrict__ wl = reinterpret_cast<const v16bf*>(g_Wlo);

    v8f acc[4];
#pragma unroll
    for (int t = 0; t < 4; ++t)
#pragma unroll
        for (int r = 0; r < 8; ++r) acc[t][r] = 0.0f;

    const float* xbase = x + (size_t)(tb + ln) * D_MODEL + half * 8;

    for (int kc = 0; kc < KCHUNKS; ++kc) {
        // --- A fragment: 16x32 fp32 -> bf16 hi/lo (truncating split).
        //     lane M = ln; elem e -> K = (e>=8?16:0) + half*8 + (e&7)
        const float* xr = xbase + kc * 32;
        __builtin_prefetch((const void*)(xr + 64), 0, 1);   // global_prefetch_b8
        float4 a0 = *reinterpret_cast<const float4*>(xr);
        float4 a1 = *reinterpret_cast<const float4*>(xr + 4);
        float4 a2 = *reinterpret_cast<const float4*>(xr + 16);
        float4 a3 = *reinterpret_cast<const float4*>(xr + 20);
        float fa[16] = {a0.x, a0.y, a0.z, a0.w, a1.x, a1.y, a1.z, a1.w,
                        a2.x, a2.y, a2.z, a2.w, a3.x, a3.y, a3.z, a3.w};
        v8u hw, lw;
#pragma unroll
        for (int w = 0; w < 8; ++w) {
            float fe = fa[2 * w], fo = fa[2 * w + 1];
            unsigned be = __float_as_uint(fe), bo = __float_as_uint(fo);
            unsigned he = be & 0xffff0000u, ho = bo & 0xffff0000u;
            hw[w] = ho | (be >> 16);                       // packed hi (RTZ)
            float re = fe - __uint_as_float(he);
            float ro = fo - __uint_as_float(ho);
            lw[w] = (__float_as_uint(ro) & 0xffff0000u) |
                    (__float_as_uint(re) >> 16);           // packed lo (RTZ)
        }
        v16bf ahi = __builtin_bit_cast(v16bf, hw);
        v16bf alo = __builtin_bit_cast(v16bf, lw);

#pragma unroll
        for (int t = 0; t < 4; ++t) {
            int fi = (kc * 4 + t) * 32 + lane;             // v16bf index
            v16bf bhi = wh[fi];
            v16bf blo = wl[fi];
            // fp32-emulating product: ahi*bhi + alo*bhi + ahi*blo
            acc[t] = __builtin_amdgcn_wmma_f32_16x16x32_bf16(
                false, ahi, false, bhi, (short)0, acc[t], false, false);
            acc[t] = __builtin_amdgcn_wmma_f32_16x16x32_bf16(
                false, alo, false, bhi, (short)0, acc[t], false, false);
            acc[t] = __builtin_amdgcn_wmma_f32_16x16x32_bf16(
                false, ahi, false, blo, (short)0, acc[t], false, false);
        }
    }

    // --- bias; C/D layout: elem r -> token M = r + 8*half, expert N = t*16+ln
    float l[4][8];
#pragma unroll
    for (int t = 0; t < 4; ++t) {
        float bv = b[t * 16 + ln];
#pragma unroll
        for (int r = 0; r < 8; ++r) l[t][r] = acc[t][r] + bv;
    }

    // --- softmax over 64 experts (4 tiles x 16 lanes within a half) + top-2
    float impl[4] = {0.f, 0.f, 0.f, 0.f};
#pragma unroll
    for (int r = 0; r < 8; ++r) {
        float m = l[0][r];
#pragma unroll
        for (int t = 1; t < 4; ++t) m = fmaxf(m, l[t][r]);
#pragma unroll
        for (int d = 1; d < 16; d <<= 1) m = fmaxf(m, __shfl_xor(m, d, 32));

        float p[4]; float s = 0.f;
#pragma unroll
        for (int t = 0; t < 4; ++t) { p[t] = __expf(l[t][r] - m); s += p[t]; }
#pragma unroll
        for (int d = 1; d < 16; d <<= 1) s += __shfl_xor(s, d, 32);
        float inv = 1.0f / s;

        float g1 = -1.f, g2 = -1.f; int e1 = 1 << 30, e2 = 1 << 30;
#pragma unroll
        for (int t = 0; t < 4; ++t) {
            float pt = p[t] * inv;
            impl[t] += pt;
            int ex = t * 16 + ln;
            if (beat(pt, ex, g1, e1)) { g2 = g1; e2 = e1; g1 = pt; e1 = ex; }
            else if (beat(pt, ex, g2, e2)) { g2 = pt; e2 = ex; }
        }
        // butterfly top-2 merge across the 16-lane half
#pragma unroll
        for (int d = 1; d < 16; d <<= 1) {
            float og1 = __shfl_xor(g1, d, 32); int oe1 = __shfl_xor(e1, d, 32);
            float og2 = __shfl_xor(g2, d, 32); int oe2 = __shfl_xor(e2, d, 32);
            if (beat(og1, oe1, g1, e1)) {
                if (beat(g1, e1, og2, oe2)) { g2 = g1; e2 = e1; }
                else { g2 = og2; e2 = oe2; }
                g1 = og1; e1 = oe1;
            } else if (beat(og1, oe1, g2, e2)) { g2 = og1; e2 = oe1; }
        }
        if (ln == 0) {
            int token = tb + half * 8 + r;
            out_ids[token * 2 + 0]   = (float)e1;
            out_ids[token * 2 + 1]   = (float)e2;
            out_gates[token * 2 + 0] = g1;
            out_gates[token * 2 + 1] = g2;
        }
    }

    // importance[e] += sum of probs (full softmax, all experts)
#pragma unroll
    for (int t = 0; t < 4; ++t) {
        float v = impl[t];
        v += __shfl_xor(v, 16, 32);          // combine the two halves
        if (half == 0) atomicAdd(&imp[t * 16 + ln], v);
    }
}

// ---------------------------------------------------------------------------
// K2: per-expert capacity drop. One block per expert; compact slots into LDS,
// rank by (gate desc, index asc), keep rank < cap. load[e] = min(count, cap).
// ---------------------------------------------------------------------------
__global__ __launch_bounds__(256) void moe_capacity_kernel(
    const float* __restrict__ ids, const float* __restrict__ gates,
    float* __restrict__ mask, float* __restrict__ loadv, int M2, int cap)
{
    __shared__ float gb[LC];
    __shared__ int   ib[LC];
    __shared__ int   scnt;
    const int e = blockIdx.x;
    if (threadIdx.x == 0) scnt = 0;
    __syncthreads();

    for (int i = threadIdx.x; i < M2; i += 256) {
        int id = (int)(ids[i] + 0.5f);
        if (id == e) {
            int p = atomicAdd(&scnt, 1);
            if (p < LC) { gb[p] = gates[i]; ib[p] = i; }
        }
    }
    __syncthreads();
    int C = scnt;
    if (threadIdx.x == 0) loadv[e] = (float)(C < cap ? C : cap);

    if (C <= cap) {
        for (int j = threadIdx.x; j < C; j += 256) mask[ib[j]] = 1.0f;
    } else if (C <= LC) {
        for (int j = threadIdx.x; j < C; j += 256) {
            float gj = gb[j]; int ij = ib[j]; int rank = 0;
            for (int k = 0; k < C; ++k)
                rank += (gb[k] > gj) || (gb[k] == gj && ib[k] < ij);
            if (rank < cap) mask[ij] = 1.0f;
        }
    } else {    // pathological skew: rank via global re-scan (L2-resident)
        for (int i = threadIdx.x; i < M2; i += 256) {
            if ((int)(ids[i] + 0.5f) != e) continue;
            float gi = gates[i]; int rank = 0;
            for (int j = 0; j < M2; ++j) {
                if ((int)(ids[j] + 0.5f) != e) continue;
                float gj = gates[j];
                rank += (gj > gi) || (gj == gi && j < i);
            }
            if (rank < cap) mask[i] = 1.0f;
        }
    }
}

// ---------------------------------------------------------------------------
// K3: aux loss = sum_e  E * (importance_e / N) * (load_e / N)
// ---------------------------------------------------------------------------
__global__ void moe_aux_kernel(const float* __restrict__ ws, float* __restrict__ out,
                               int N, int NK)
{
    __shared__ float red[N_EXP];
    int e = threadIdx.x;
    float invN = 1.0f / (float)N;
    red[e] = (float)N_EXP * (ws[e] * invN) * (ws[N_EXP + e] * invN);
    __syncthreads();
    if (e == 0) {
        float s = 0.f;
        for (int i = 0; i < N_EXP; ++i) s += red[i];
        out[2 * NK] = s;
    }
}

// ---------------------------------------------------------------------------
extern "C" void kernel_launch(void* const* d_in, const int* in_sizes, int n_in,
                              void* d_out, int out_size, void* d_ws, size_t ws_size,
                              hipStream_t stream) {
    const float* x = (const float*)d_in[0];
    const float* W = (const float*)d_in[1];
    const float* b = (const float*)d_in[2];
    // d_in[3] = training flag (eval path only) — unused

    const int N  = in_sizes[0] / D_MODEL;     // 16384
    const int NK = N * TOPK;                  // 32768
    const int cap = (int)ceil(1.2 * (double)N / (double)N_EXP);   // 308

    float* outv      = (float*)d_out;
    float* out_ids   = outv;                  // [NK]  (ids as float)
    float* out_gates = outv + NK;             // [NK]
    float* out_mask  = outv + 2 * NK + 1;     // [NK]  (aux scalar at 2*NK)

    float* imp   = (float*)d_ws;              // [64] importance accumulators
    float* loadv = imp + N_EXP;               // [64] load per expert

    moe_wpack_kernel<<<(WFRAG_N + 255) / 256, 256, 0, stream>>>(W);
    moe_init_kernel<<<(NK + 255) / 256, 256, 0, stream>>>(out_mask, imp, NK);
    moe_router_gemm_kernel<<<N / 128, 256, 0, stream>>>(x, b, out_ids,
                                                        out_gates, imp, N);
    moe_capacity_kernel<<<N_EXP, 256, 0, stream>>>(out_ids, out_gates, out_mask,
                                                   loadv, NK, cap);
    moe_aux_kernel<<<1, N_EXP, 0, stream>>>(imp, outv, N, NK);
}
